// NearestNeighbours_37417755083180
// MI455X (gfx1250) — matrix operand order repaired and checked by reading, not compiled, not run on hardware
//
#include <hip/hip_runtime.h>

// ---------------- types ----------------
typedef __attribute__((ext_vector_type(16))) __bf16 v16bf;
typedef __attribute__((ext_vector_type(8)))  __bf16 v8bf;
typedef __attribute__((ext_vector_type(8)))  float  v8f;

#define EDIM    512
#define VROWS   50000
#define QROWS   4096          // B*S = 8*512
#define VTILES  (VROWS / 16)  // 3125, exact
#define VSPLIT  16            // V-dimension slices -> grid = 32 x 16 = 512 blocks
#define TILE_BYTES 16384      // 16 vocab rows x 512 K x 2B

// ---------------------------------------------------------------
// Row-wise L2 normalize fp32 -> bf16.  One wave per row.
// ---------------------------------------------------------------
__global__ __launch_bounds__(256)
void rownorm_bf16(const float* __restrict__ in, __bf16* __restrict__ out, int rows)
{
    const int wid  = threadIdx.x >> 5;
    const int lane = threadIdx.x & 31;
    const int row  = blockIdx.x * (blockDim.x >> 5) + wid;
    if (row >= rows) return;

    const float* r = in + (size_t)row * EDIM + lane * 16;
    float x[16];
    float ss = 0.0f;
#pragma unroll
    for (int i = 0; i < 4; ++i) {
        float4 v = *(const float4*)(r + i * 4);
        x[i*4+0] = v.x; x[i*4+1] = v.y; x[i*4+2] = v.z; x[i*4+3] = v.w;
        ss += v.x*v.x + v.y*v.y + v.z*v.z + v.w*v.w;
    }
#pragma unroll
    for (int m = 16; m >= 1; m >>= 1)
        ss += __shfl_xor(ss, m, 32);

    const float inv = 1.0f / sqrtf(ss);

    v16bf o;
#pragma unroll
    for (int i = 0; i < 16; ++i)
        o[i] = (__bf16)(x[i] * inv);

    *(v16bf*)(out + (size_t)row * EDIM + lane * 16) = o;
}

// ---------------------------------------------------------------
// init packed-argmax array
// ---------------------------------------------------------------
__global__ void init_pk(unsigned long long* pk, int n)
{
    int i = blockIdx.x * blockDim.x + threadIdx.x;
    if (i < n) pk[i] = 0ull;
}

// ---------------------------------------------------------------
// Main kernel: 8 waves/block, wave w owns 16 queries (B register-resident,
// 128 VGPRs).  Vocab tiles (16 rows x 512 K, 16KB bf16) are shared by all
// waves via double-buffered LDS.  Global loads for tile i+1 issue before the
// WMMA chain on tile i.  A operands are pipelined from LDS in groups of 4
// chunks with hard sched_barriers: loads(g+1) -> barrier -> 4x WMMA(g),
// so each WMMA batch waits dscnt<=8, never 0, with no register recycling.
// ---------------------------------------------------------------
__global__ __launch_bounds__(256, 1)
void nn_argmax_wmma(const __bf16* __restrict__ embN,
                    const __bf16* __restrict__ qN,
                    unsigned long long* __restrict__ pk)
{
    __shared__ __align__(16) char smem[2 * TILE_BYTES];

    const int tid  = threadIdx.x;
    const int wid  = tid >> 5;           // 0..7
    const int lane = tid & 31;
    const int half = lane >> 4;          // 0 | 1
    const int l16  = lane & 15;
    const int qg   = blockIdx.x;         // 0..31  (query group of 128)
    const int vs   = blockIdx.y;         // 0..VSPLIT-1
    const int qrow = qg * 128 + wid * 16 + l16;

    // ---- resident B matrix (queries), 32x16 bf16, contiguous k = c*32 + 16*half + s ----
    v16bf bq[16];
    {
        const __bf16* qb = qN + (size_t)qrow * EDIM + half * 16;
#pragma unroll
        for (int c = 0; c < 16; ++c)
            bq[c] = *(const v16bf*)(qb + c * 32);
    }

    const int t0 = (VTILES * vs)       / VSPLIT;
    const int t1 = (VTILES * (vs + 1)) / VSPLIT;
    const int ntiles = t1 - t0;

    const char* gsrc = (const char*)embN;

    // stage first tile into buffer 0 (thread t copies 64 contiguous bytes)
    {
        const char* s = gsrc + (size_t)t0 * TILE_BYTES + tid * 64;
        int4 r0 = *(const int4*)(s);
        int4 r1 = *(const int4*)(s + 16);
        int4 r2 = *(const int4*)(s + 32);
        int4 r3 = *(const int4*)(s + 48);
        char* d = smem + tid * 64;
        *(int4*)(d)      = r0;
        *(int4*)(d + 16) = r1;
        *(int4*)(d + 32) = r2;
        *(int4*)(d + 48) = r3;
    }
    __syncthreads();

    float bestSim = -__builtin_inff();
    int   bestIdx = 0;

    for (int i = 0; i < ntiles; ++i) {
        const int  buf  = i & 1;
        const bool more = (i + 1) < ntiles;    // block-uniform

        // issue global loads for next tile (consumed after compute)
        int4 r0 = {}, r1 = {}, r2 = {}, r3 = {};
        if (more) {
            const char* s = gsrc + (size_t)(t0 + i + 1) * TILE_BYTES + tid * 64;
            r0 = *(const int4*)(s);
            r1 = *(const int4*)(s + 16);
            r2 = *(const int4*)(s + 32);
            r3 = *(const int4*)(s + 48);
        }

        // ---- WMMA chain on current LDS buffer ----
        // A lane layout: lanes0-15 row=l16, K{c*32+0..15};
        //                lanes16-31 row=l16, K{+16..31}
        const char* lb = smem + buf * TILE_BYTES + l16 * 1024 + half * 32;

        // double-buffered groups of 4 chunks (2 x 32 VGPRs)
        v8bf alo[2][4], ahi[2][4];
#pragma unroll
        for (int j = 0; j < 4; ++j) {
            alo[0][j] = *(const v8bf*)(lb + j * 64);
            ahi[0][j] = *(const v8bf*)(lb + j * 64 + 16);
        }

        v8f acc = {};
#pragma unroll
        for (int g = 0; g < 4; ++g) {
            const int cur = g & 1;
            const int nxt = cur ^ 1;
            if (g < 3) {                        // issue loads for group g+1
#pragma unroll
                for (int j = 0; j < 4; ++j) {
                    alo[nxt][j] = *(const v8bf*)(lb + ((g + 1) * 4 + j) * 64);
                    ahi[nxt][j] = *(const v8bf*)(lb + ((g + 1) * 4 + j) * 64 + 16);
                }
            }
            __builtin_amdgcn_sched_barrier(0);  // loads(g+1) pinned above
#pragma unroll
            for (int j = 0; j < 4; ++j) {       // 4 WMMAs on group g
                v16bf a = __builtin_shufflevector(alo[cur][j], ahi[cur][j],
                            0,1,2,3,4,5,6,7,8,9,10,11,12,13,14,15);
                acc = __builtin_amdgcn_wmma_f32_16x16x32_bf16(
                          false, a, false, bq[g * 4 + j], (short)0, acc,
                          false, false);
            }
            __builtin_amdgcn_sched_barrier(0);  // wmmas(g) pinned above
        }

        // per-lane argmax: all 8 rows belong to query (lane&15)
        const int vt = t0 + i;
#pragma unroll
        for (int g = 0; g < 8; ++g) {
            const float s = acc[g];
            if (s > bestSim) {                  // '>' keeps first (lowest) index
                bestSim = s;
                bestIdx = vt * 16 + g + half * 8;
            }
        }

        // commit next tile into the spare buffer
        if (more) {
            char* d = smem + (buf ^ 1) * TILE_BYTES + tid * 64;
            *(int4*)(d)      = r0;
            *(int4*)(d + 16) = r1;
            *(int4*)(d + 32) = r2;
            *(int4*)(d + 48) = r3;
        }
        __syncthreads();
    }

    // merge the two lane-halves (both track query lane&15)
    {
        const float os = __shfl_xor(bestSim, 16, 32);
        const int   oi = __shfl_xor(bestIdx, 16, 32);
        if (os > bestSim || (os == bestSim && oi < bestIdx)) {
            bestSim = os; bestIdx = oi;
        }
    }

    // merge across V-slices: packed (ordered_sim | ~idx) atomic max
    if (lane < 16) {
        unsigned int b   = __float_as_uint(bestSim);
        unsigned int ord = (b & 0x80000000u) ? ~b : (b | 0x80000000u);
        unsigned long long key =
            ((unsigned long long)ord << 32) | (unsigned int)(~bestIdx);
        atomicMax(pk + qrow, key);
    }
}

// ---------------------------------------------------------------
// unpack argmax index -> float output
// ---------------------------------------------------------------
__global__ void finalize_idx(const unsigned long long* __restrict__ pk,
                             float* __restrict__ out, int n)
{
    int i = blockIdx.x * blockDim.x + threadIdx.x;
    if (i < n)
        out[i] = (float)(~(unsigned int)(pk[i] & 0xffffffffu));
}

// ---------------------------------------------------------------
// launcher
// ---------------------------------------------------------------
extern "C" void kernel_launch(void* const* d_in, const int* in_sizes, int n_in,
                              void* d_out, int out_size, void* d_ws, size_t ws_size,
                              hipStream_t stream)
{
    (void)in_sizes; (void)n_in; (void)out_size; (void)ws_size;

    const float* batch = (const float*)d_in[0];   // [8,512,512] fp32
    const float* emb   = (const float*)d_in[1];   // [50000,512] fp32

    // workspace layout (bytes): embN bf16 | qN bf16 | packed argmax
    __bf16* embN = (__bf16*)d_ws;
    __bf16* qN   = (__bf16*)((char*)d_ws + (size_t)VROWS * EDIM * 2);
    unsigned long long* pk =
        (unsigned long long*)((char*)d_ws + (size_t)VROWS * EDIM * 2
                                          + (size_t)QROWS * EDIM * 2);

    rownorm_bf16<<<(VROWS + 7) / 8, 256, 0, stream>>>(emb, embN, VROWS);
    rownorm_bf16<<<(QROWS + 7) / 8, 256, 0, stream>>>(batch, qN, QROWS);
    init_pk<<<(QROWS + 255) / 256, 256, 0, stream>>>(pk, QROWS);

    dim3 grid(QROWS / 128, VSPLIT);
    nn_argmax_wmma<<<grid, 256, 0, stream>>>(embN, qN, pk);

    finalize_idx<<<(QROWS + 255) / 256, 256, 0, stream>>>(pk, (float*)d_out, QROWS);
}
